// ExampleTGCN_82755429859488
// MI455X (gfx1250) — matrix operand order repaired and checked by reading, not compiled, not run on hardware
//
#include <hip/hip_runtime.h>
#include <hip/hip_bf16.h>

// ---------------------------------------------------------------------------
// TGCN (3 timesteps) on MI455X / gfx1250, wave32 + WMMA f16->f32.
//
//  - All dense GEMMs use v_wmma_f32_16x16x32_f16 (f32 accumulate).
//  - B (weight) matrices pre-packed once into WMMA fragment order:
//      fragment load = 2x global_load_b128 per lane (coalesced, L2-resident).
//  - A (activation) tile staged in LDS per block (f32->f16 once), fragments
//    read as 2x ds_load_b128 (layout has 8 contiguous K per half-group).
//  - Edge aggregation: 128 lanes = one edge row -> coalesced gather +
//    coalesced global_atomic_add_f32 (the HBM-bound part, ~16 GB total).
// ---------------------------------------------------------------------------

#define NN   100000
#define EE   1600000
#define DHF  128
#define PB   500      // max-pool partial blocks (100000 / 500 = 200 rows each)

typedef __attribute__((ext_vector_type(16))) _Float16 v16h;
typedef __attribute__((ext_vector_type(8)))  _Float16 v8h;
typedef __attribute__((ext_vector_type(8)))  float    v8f;
typedef __attribute__((ext_vector_type(4)))  float    v4f;

// ---------------------------------------------------------------------------
// Pack one 128x128 f32 matrix (row-major, [K x N]) into WMMA B-fragment order:
//   dst[((ntile*4 + kt)*32 + lane)*16 + e] = src[(kt*32 + (lane>>4)*8 + (e&7) + (e>>3)*16)*128
//                                                + ntile*16 + (lane&15)]
// One thread per destination half (16384 total).
__global__ void pack_b_f16(const float* __restrict__ src, _Float16* __restrict__ dst) {
    int t = blockIdx.x * blockDim.x + threadIdx.x;
    if (t >= 16384) return;
    int e    = t & 15;
    int lane = (t >> 4) & 31;
    int kt   = (t >> 9) & 3;
    int nt   = t >> 11;
    int col  = nt * 16 + (lane & 15);
    int K    = kt * 32 + ((lane >> 4) << 3) + (e & 7) + ((e >> 3) << 4);
    dst[t] = (_Float16)src[K * DHF + col];
}

// ---------------------------------------------------------------------------
// C[M x 128] = A1[M x 128]@B1 (+ A2@B2) + bias.  B pre-packed (see above).
// grid.x = M/16 row tiles; block = 256 thr = 8 wave32; wave w -> col tile w.
__global__ void __launch_bounds__(256)
gemm128_wmma(const float* __restrict__ A1, const _Float16* __restrict__ B1p,
             const float* __restrict__ A2, const _Float16* __restrict__ B2p,
             const float* __restrict__ bias, float* __restrict__ C)
{
    __shared__ _Float16 As[16 * DHF];          // 4 KB: 16 rows x 128 halves

    int tid  = threadIdx.x;
    int lane = tid & 31;
    int wave = tid >> 5;
    int m0   = blockIdx.x << 4;

    // cooperative A-tile stage: thread -> (row = tid>>4, K chunk of 8)
    int lrow = tid >> 4;
    int kc   = (tid & 15) << 3;
    {
        const float* p = A1 + (m0 + lrow) * DHF + kc;
        v4f f0 = *(const v4f*)(p);
        v4f f1 = *(const v4f*)(p + 4);
        v8h h;
#pragma unroll
        for (int i = 0; i < 4; ++i) { h[i] = (_Float16)f0[i]; h[4 + i] = (_Float16)f1[i]; }
        *(v8h*)(&As[lrow * DHF + kc]) = h;
    }
    __syncthreads();

    int rowA = lane & 15;                      // A frag: M = lane&15
    int kA   = (lane >> 4) << 3;               // +0 / +8 K group
    int colB = (wave << 4) + (lane & 15);
    const _Float16* bp = B1p + ((size_t)(wave * 4) * 32 + lane) * 16;

    v8f acc = {};
#pragma unroll
    for (int kt = 0; kt < 4; ++kt) {
        int kbase = (kt << 5) + kA;
        v8h alo = *(const v8h*)(&As[rowA * DHF + kbase]);        // K = kbase..+7
        v8h ahi = *(const v8h*)(&As[rowA * DHF + kbase + 16]);   // K = kbase+16..+23
        v16h a;
#pragma unroll
        for (int i = 0; i < 8; ++i) { a[i] = alo[i]; a[8 + i] = ahi[i]; }
        const _Float16* bq = bp + kt * 32 * 16;
        v8h b0 = *(const v8h*)(bq);
        v8h b1 = *(const v8h*)(bq + 8);
        v16h b;
#pragma unroll
        for (int i = 0; i < 8; ++i) { b[i] = b0[i]; b[8 + i] = b1[i]; }
        acc = __builtin_amdgcn_wmma_f32_16x16x32_f16(false, a, false, b, (short)0, acc, false, false);
    }

    if (A2) {                                  // uniform branch
        __syncthreads();                       // all waves done reading As
        {
            const float* p = A2 + (m0 + lrow) * DHF + kc;
            v4f f0 = *(const v4f*)(p);
            v4f f1 = *(const v4f*)(p + 4);
            v8h h;
#pragma unroll
            for (int i = 0; i < 4; ++i) { h[i] = (_Float16)f0[i]; h[4 + i] = (_Float16)f1[i]; }
            *(v8h*)(&As[lrow * DHF + kc]) = h;
        }
        __syncthreads();
        const _Float16* bp2 = B2p + ((size_t)(wave * 4) * 32 + lane) * 16;
#pragma unroll
        for (int kt = 0; kt < 4; ++kt) {
            int kbase = (kt << 5) + kA;
            v8h alo = *(const v8h*)(&As[rowA * DHF + kbase]);
            v8h ahi = *(const v8h*)(&As[rowA * DHF + kbase + 16]);
            v16h a;
#pragma unroll
            for (int i = 0; i < 8; ++i) { a[i] = alo[i]; a[8 + i] = ahi[i]; }
            const _Float16* bq = bp2 + kt * 32 * 16;
            v8h b0 = *(const v8h*)(bq);
            v8h b1 = *(const v8h*)(bq + 8);
            v16h b;
#pragma unroll
            for (int i = 0; i < 8; ++i) { b[i] = b0[i]; b[8 + i] = b1[i]; }
            acc = __builtin_amdgcn_wmma_f32_16x16x32_f16(false, a, false, b, (short)0, acc, false, false);
        }
    }

    // C/D layout: VGPR v -> row m0 + (lane>>4)*8 + v, col = colB
    float bv = bias ? bias[colB] : 0.0f;
    int rbase = m0 + ((lane >> 4) << 3);
#pragma unroll
    for (int v = 0; v < 8; ++v)
        C[(rbase + v) * DHF + colB] = acc[v] + bv;
}

// ---------------- small utility kernels ----------------
__global__ void zero_f32(float* p, int n) {
    int i = blockIdx.x * blockDim.x + threadIdx.x;
    if (i < n) p[i] = 0.0f;
}
__global__ void deg_init(float* deg, int n) {
    int i = blockIdx.x * blockDim.x + threadIdx.x;
    if (i < n) deg[i] = 1.0f;                    // self-loop
}
__global__ void deg_scatter(const int* __restrict__ dst, float* deg, int e) {
    int i = blockIdx.x * blockDim.x + threadIdx.x;
    if (i < e) atomicAdd(&deg[dst[i]], 1.0f);
}
__global__ void dinv_k(const float* __restrict__ deg, float* dinv, int n) {
    int i = blockIdx.x * blockDim.x + threadIdx.x;
    if (i < n) dinv[i] = rsqrtf(deg[i]);
}
__global__ void enorm_k(const int* __restrict__ src, const int* __restrict__ dst,
                        const float* __restrict__ dinv, float* enorm, int e) {
    int i = blockIdx.x * blockDim.x + threadIdx.x;
    if (i < e) enorm[i] = dinv[src[i]] * dinv[dst[i]];
}

// AGGg = dinv^2 * XWg + convb_g   (self-loop contribution + bias)
__global__ void agg_init(const float* __restrict__ dinv,
                         const float* __restrict__ xwz, const float* __restrict__ xwr,
                         const float* __restrict__ xwh,
                         float* aggz, float* aggr, float* aggh,
                         const float* __restrict__ bz, const float* __restrict__ br,
                         const float* __restrict__ bh)
{
    int t = blockIdx.x * blockDim.x + threadIdx.x;
    int i = t >> 7, f = t & 127;
    if (i >= NN) return;
    float d2 = dinv[i] * dinv[i];
    int idx = i * DHF + f;
    aggz[idx] = d2 * xwz[idx] + bz[f];
    aggr[idx] = d2 * xwr[idx] + br[f];
    aggh[idx] = d2 * xwh[idx] + bh[f];
}

// Coalesced edge scatter: 128 consecutive lanes = one edge's feature row.
__global__ void scatter_edges(const int* __restrict__ src, const int* __restrict__ dst,
                              const float* __restrict__ enorm,
                              const float* __restrict__ xwz, const float* __restrict__ xwr,
                              const float* __restrict__ xwh,
                              float* aggz, float* aggr, float* aggh)
{
    int t = blockIdx.x * blockDim.x + threadIdx.x;
    int e = t >> 7, f = t & 127;
    if (e >= EE) return;
    int s = src[e], d = dst[e];
    float w = enorm[e];
    int si = s * DHF + f, di = d * DHF + f;
    atomicAdd(&aggz[di], w * xwz[si]);
    atomicAdd(&aggr[di], w * xwr[si]);
    atomicAdd(&aggh[di], w * xwh[si]);
}

// Zs = sigmoid(Zpre) in-place; Rpre <- H * sigmoid(Rpre) in-place
__global__ void gates_zr(float* zpre, float* rpre, const float* __restrict__ H, int n) {
    int i = blockIdx.x * blockDim.x + threadIdx.x;
    if (i >= n) return;
    zpre[i] = 1.0f / (1.0f + __expf(-zpre[i]));
    float rs = 1.0f / (1.0f + __expf(-rpre[i]));
    rpre[i] = H[i] * rs;
}
// H = Z*H + (1-Z)*tanh(Hpre)
__global__ void gates_h(float* H, const float* __restrict__ zs, const float* __restrict__ hpre, int n) {
    int i = blockIdx.x * blockDim.x + threadIdx.x;
    if (i >= n) return;
    float z = zs[i];
    H[i] = z * H[i] + (1.0f - z) * tanhf(hpre[i]);
}

// ---------------- pooling + final GEMV ----------------
__global__ void pool_partial(const float* __restrict__ H, float* part, int rowsPerBlk) {
    int f = threadIdx.x;
    int r0 = blockIdx.x * rowsPerBlk;
    int r1 = r0 + rowsPerBlk; if (r1 > NN) r1 = NN;
    float m = -3.4e38f;
    for (int r = r0; r < r1; ++r) m = fmaxf(m, H[r * DHF + f]);
    part[blockIdx.x * DHF + f] = m;
}
__global__ void pool_final(const float* __restrict__ part, int nb,
                           const float* __restrict__ w2, const float* __restrict__ b2,
                           float* __restrict__ out)
{
    __shared__ float h[DHF];
    int f = threadIdx.x;
    float m = -3.4e38f;
    for (int b = 0; b < nb; ++b) m = fmaxf(m, part[b * DHF + f]);
    h[f] = m;
    __syncthreads();
    if (f < 10) {
        float acc = b2[f];
        for (int k = 0; k < DHF; ++k) acc += h[k] * w2[k * 10 + f];
        out[f] = acc;
    }
}

// ---------------------------------------------------------------------------
extern "C" void kernel_launch(void* const* d_in, const int* in_sizes, int n_in,
                              void* d_out, int out_size, void* d_ws, size_t ws_size,
                              hipStream_t stream)
{
    // Inputs (setup_inputs dict order)
    const float* X[3]    = { (const float*)d_in[0], (const float*)d_in[2], (const float*)d_in[4] };
    const int*   EDG[3]  = { (const int*)d_in[1],   (const int*)d_in[3],   (const int*)d_in[5]   };
    const float* lin1_w  = (const float*)d_in[6];
    const float* lin1_b  = (const float*)d_in[7];
    const float* convW[3] = { (const float*)d_in[8],  (const float*)d_in[12], (const float*)d_in[16] };
    const float* convb[3] = { (const float*)d_in[9],  (const float*)d_in[13], (const float*)d_in[17] };
    const float* linW[3]  = { (const float*)d_in[10], (const float*)d_in[14], (const float*)d_in[18] };
    const float* linb[3]  = { (const float*)d_in[11], (const float*)d_in[15], (const float*)d_in[19] };
    const float* lin2_w  = (const float*)d_in[20];
    const float* lin2_b  = (const float*)d_in[21];
    float* out = (float*)d_out;

    // Workspace layout (~420 MB)
    const size_t NF = (size_t)NN * DHF;
    float* ws    = (float*)d_ws;
    float* H     = ws;
    float* XI    = H + NF;
    float* XW[3]  = { XI + NF, XI + 2 * NF, XI + 3 * NF };
    float* AGG[3] = { XI + 4 * NF, XI + 5 * NF, XI + 6 * NF };
    float* deg   = XI + 7 * NF;
    float* dinv  = deg + NN;
    float* enorm = dinv + NN;
    float* part  = enorm + EE;
    _Float16* lin1h = (_Float16*)(part + (size_t)PB * DHF);  // packed fragment order
    _Float16* convh = lin1h + 16384;            // 3 x 128x128 packed
    _Float16* linh  = convh + 3 * 16384;        // 3 x (top 128x128 | bottom 128x128) packed

    const int nfi = (int)NF;                    // 12.8M, fits int
    const int TPB = 256;
    const int gridNF   = (nfi + TPB - 1) / TPB;
    const int gridN    = (NN + TPB - 1) / TPB;
    const int gridE    = (EE + TPB - 1) / TPB;
    const int gridEF   = (EE * DHF) / TPB;      // 800000
    const int gridGEMM = NN / 16;               // 6250
    const int gridPK   = 16384 / TPB;           // 64

    // Pre-pack all weight matrices into WMMA fragment layout (once per launch)
    pack_b_f16<<<gridPK, TPB, 0, stream>>>(lin1_w, lin1h);
    for (int g = 0; g < 3; ++g) {
        pack_b_f16<<<gridPK, TPB, 0, stream>>>(convW[g], convh + g * 16384);
        pack_b_f16<<<gridPK, TPB, 0, stream>>>(linW[g],          linh + g * 32768);          // top rows 0..127
        pack_b_f16<<<gridPK, TPB, 0, stream>>>(linW[g] + 16384,  linh + g * 32768 + 16384);  // bottom rows 128..255
    }

    // H = 0
    zero_f32<<<gridNF, TPB, 0, stream>>>(H, nfi);

    for (int t = 0; t < 3; ++t) {
        const int* src = EDG[t];
        const int* dst = EDG[t] + EE;

        // degree / norm
        deg_init<<<gridN, TPB, 0, stream>>>(deg, NN);
        deg_scatter<<<gridE, TPB, 0, stream>>>(dst, deg, EE);
        dinv_k<<<gridN, TPB, 0, stream>>>(deg, dinv, NN);
        enorm_k<<<gridE, TPB, 0, stream>>>(src, dst, dinv, enorm, EE);

        // XI = X @ lin1 + b1
        gemm128_wmma<<<gridGEMM, TPB, 0, stream>>>(X[t], lin1h, (const float*)nullptr,
                                                   (const _Float16*)nullptr, lin1_b, XI);
        // XWg = XI @ convWg (bias added post-aggregation)
        for (int g = 0; g < 3; ++g)
            gemm128_wmma<<<gridGEMM, TPB, 0, stream>>>(XI, convh + g * 16384, (const float*)nullptr,
                                                       (const _Float16*)nullptr, (const float*)nullptr, XW[g]);
        // AGG init (self-loop + bias) then edge scatter
        agg_init<<<gridNF, TPB, 0, stream>>>(dinv, XW[0], XW[1], XW[2],
                                             AGG[0], AGG[1], AGG[2],
                                             convb[0], convb[1], convb[2]);
        scatter_edges<<<gridEF, TPB, 0, stream>>>(src, dst, enorm,
                                                  XW[0], XW[1], XW[2],
                                                  AGG[0], AGG[1], AGG[2]);

        // Gate GEMMs: concat([A, H]) @ W == A @ Wtop + H @ Wbot.  Reuse XW buffers.
        gemm128_wmma<<<gridGEMM, TPB, 0, stream>>>(AGG[0], linh + 0 * 32768, H, linh + 0 * 32768 + 16384,
                                                   linb[0], XW[0]);   // Zpre
        gemm128_wmma<<<gridGEMM, TPB, 0, stream>>>(AGG[1], linh + 1 * 32768, H, linh + 1 * 32768 + 16384,
                                                   linb[1], XW[1]);   // Rpre
        gates_zr<<<gridNF, TPB, 0, stream>>>(XW[0], XW[1], H, nfi);   // XW0<-Z, XW1<-H*R
        gemm128_wmma<<<gridGEMM, TPB, 0, stream>>>(AGG[2], linh + 2 * 32768, XW[1], linh + 2 * 32768 + 16384,
                                                   linb[2], XW[2]);   // Hpre
        gates_h<<<gridNF, TPB, 0, stream>>>(H, XW[0], XW[2], nfi);    // H update
    }

    // global max pool + final linear
    pool_partial<<<PB, DHF, 0, stream>>>(H, part, NN / PB);
    pool_final<<<1, DHF, 0, stream>>>(part, PB, lin2_w, lin2_b, out);
}
